// LMUEncoder_18219251270406
// MI455X (gfx1250) — compile-verified, matrix operand order checked by reading, not run on hardware
//
#include <hip/hip_runtime.h>
#include <hip/hip_bf16.h>
#include <stdint.h>

// ---------------- CDNA5 WMMA plumbing (gfx1250, wave32) ----------------
typedef __attribute__((ext_vector_type(16))) __bf16 v16bf;
typedef __attribute__((ext_vector_type(8)))  float  v8f;
typedef __attribute__((ext_vector_type(8)))  unsigned int v8u;

__device__ __forceinline__ uint16_t f2bf(float f) {
  uint32_t u = __builtin_bit_cast(uint32_t, f);
  u += 0x7FFFu + ((u >> 16) & 1u);          // round-to-nearest-even
  return (uint16_t)(u >> 16);
}
__device__ __forceinline__ float bf2f(uint16_t b) {
  uint32_t u = ((uint32_t)b) << 16;
  return __builtin_bit_cast(float, u);
}

// Load one 16x32 bf16 operand tile (row-major, `ld` elements per row).
// ISA 7.12.2 16-bit A/B layout: lanes 0-15 hold rows 0-15 with K = {kb..kb+7, kb+16..kb+23},
// lanes 16-31 same rows with K shifted by 8. Two b128 loads per lane.
__device__ __forceinline__ v16bf load_tile(const uint16_t* __restrict__ base, int ld) {
  int lane = threadIdx.x & 31;
  int m  = lane & 15;
  int kb = (lane >> 4) * 8;
  const uint16_t* p = base + (size_t)m * ld + kb;
  uint4 q0 = *(const uint4*)(p);
  uint4 q1 = *(const uint4*)(p + 16);
  v8u r;
  r[0] = q0.x; r[1] = q0.y; r[2] = q0.z; r[3] = q0.w;
  r[4] = q1.x; r[5] = q1.y; r[6] = q1.z; r[7] = q1.w;
  return __builtin_bit_cast(v16bf, r);
}

__device__ __forceinline__ v8f wmma_bf16(v16bf a, v16bf b, v8f c) {
  return __builtin_amdgcn_wmma_f32_16x16x32_bf16(false, a, false, b, (short)0, c, false, false);
}

// ---------------- Problem constants ----------------
#define B_SZ   8
#define T_SEQ  1024
#define D_IN   80
#define D_PAD  96      // pad K of input projection to a multiple of 32
#define H_DIM  512
#define MEM_N  256
#define L_NUM  4
#define NH_N   8
#define DK_N   64
#define BT     (B_SZ * T_SEQ)

// ---------------- Conversion / elementwise kernels ----------------
__global__ void k_f32_to_bf16(const float* __restrict__ in, uint16_t* __restrict__ out, int n) {
  int i = blockIdx.x * blockDim.x + threadIdx.x;
  if (i < n) out[i] = f2bf(in[i]);
}

__global__ void k_pad_convert(const float* __restrict__ in, uint16_t* __restrict__ out,
                              int rows, int kin, int kout) {
  int i = blockIdx.x * blockDim.x + threadIdx.x;
  if (i >= rows * kout) return;
  int r = i / kout, c = i % kout;
  out[i] = (c < kin) ? f2bf(in[(size_t)r * kin + c]) : (uint16_t)0;
}

// v viewed as (B,T,NH,DK) f32 -> vt bf16 (B,NH,DK,T)  (contraction dim contiguous for P@V^T)
__global__ void k_transpose_v(const float* __restrict__ v, uint16_t* __restrict__ vt) {
  int i = blockIdx.x * blockDim.x + threadIdx.x;
  if (i >= B_SZ * T_SEQ * NH_N * DK_N) return;
  int t = i % T_SEQ; int rem = i / T_SEQ;
  int d = rem % DK_N; rem /= DK_N;
  int h = rem % NH_N; int b = rem / NH_N;
  vt[i] = f2bf(v[(((size_t)(b * T_SEQ + t)) * NH_N + h) * DK_N + d]);
}

// out[row] = dot(x[row,:], w[:H])
__global__ void k_dot_rows(const float* __restrict__ x, const float* __restrict__ w,
                           float* __restrict__ out, int rows, int Hh) {
  int wave = threadIdx.x >> 5, lane = threadIdx.x & 31;
  int row = blockIdx.x * (blockDim.x >> 5) + wave;
  if (row >= rows) return;
  float s = 0.f;
  for (int c = lane; c < Hh; c += 32) s += x[(size_t)row * Hh + c] * w[c];
  for (int o = 16; o > 0; o >>= 1) s += __shfl_xor(s, o, 32);
  if (lane == 0) out[row] = s;
}

// LayerNorm over last dim, optional residual added first; one wave per row.
__global__ void k_layernorm(const float* __restrict__ x, const float* __restrict__ res,
                            const float* __restrict__ g, const float* __restrict__ bta,
                            float* __restrict__ out, int rows, int Hh) {
  int wave = threadIdx.x >> 5, lane = threadIdx.x & 31;
  int row = blockIdx.x * (blockDim.x >> 5) + wave;
  if (row >= rows) return;
  const float* xr = x + (size_t)row * Hh;
  const float* rr = res ? res + (size_t)row * Hh : nullptr;
  float s = 0.f, s2 = 0.f;
  for (int c = lane; c < Hh; c += 32) {
    float v = xr[c] + (rr ? rr[c] : 0.f);
    s += v; s2 += v * v;
  }
  for (int o = 16; o > 0; o >>= 1) { s += __shfl_xor(s, o, 32); s2 += __shfl_xor(s2, o, 32); }
  float mu  = s / Hh;
  float var = s2 / Hh - mu * mu;
  float inv = rsqrtf(var + 1e-5f);
  for (int c = lane; c < Hh; c += 32) {
    float v = xr[c] + (rr ? rr[c] : 0.f);
    out[(size_t)row * Hh + c] = (v - mu) * inv * g[c] + bta[c];
  }
}

// ---------------- expm(M) for the LMU state matrices (tiny, fp32) ----------------
__global__ void k_build_M(float* __restrict__ X, float inv2s) {
  int i = blockIdx.x * blockDim.x + threadIdx.x;
  const int n = MEM_N + 1;
  if (i >= n * n) return;
  int r = i / n, c = i % n;
  float v = 0.f;
  if (r < MEM_N) {
    float R = (2.f * r + 1.f) / 128.f;                 // THETA = 128
    if (c < MEM_N) v = (r < c) ? -R : ((((r - c) & 1) == 0) ? -R : R);
    else           v = ((r & 1) == 0) ? R : -R;        // B vector column
  }
  X[i] = v * inv2s;
}
__global__ void k_eye_plus(float* __restrict__ E, const float* __restrict__ X) {
  int i = blockIdx.x * blockDim.x + threadIdx.x;
  const int n = MEM_N + 1;
  if (i >= n * n) return;
  E[i] = X[i] + ((i / n) == (i % n) ? 1.f : 0.f);
}
__global__ void k_copy_f32(float* __restrict__ d, const float* __restrict__ s, int n) {
  int i = blockIdx.x * blockDim.x + threadIdx.x;
  if (i < n) d[i] = s[i];
}
__global__ void k_add_f32(float* __restrict__ d, const float* __restrict__ s, int n) {
  int i = blockIdx.x * blockDim.x + threadIdx.x;
  if (i < n) d[i] += s[i];
}
__global__ void k_gemm_naive(const float* __restrict__ A, const float* __restrict__ Bm,
                             float* __restrict__ C, float scale) {
  int i = blockIdx.x * blockDim.x + threadIdx.x;
  const int n = MEM_N + 1;
  if (i >= n * n) return;
  int r = i / n, c = i % n;
  float s = 0.f;
  for (int k = 0; k < n; ++k) s += A[r * n + k] * Bm[k * n + c];
  C[i] = s * scale;
}
__global__ void k_extract_AB(const float* __restrict__ E, uint16_t* __restrict__ Adbf,
                             float* __restrict__ Bd) {
  int i = blockIdx.x * blockDim.x + threadIdx.x;
  const int n = MEM_N + 1;
  if (i < MEM_N * MEM_N) {
    int r = i / MEM_N, c = i % MEM_N;
    Adbf[i] = f2bf(E[r * n + c]);
  } else if (i < MEM_N * MEM_N + MEM_N) {
    int r = i - MEM_N * MEM_N;
    Bd[r] = E[r * n + MEM_N];
  }
}

// ---------------- Main bf16 WMMA GEMM ----------------
// C[M,N] = A[M,K] @ B[N,K]^T + bias, writing f32 (Cf) and/or bf16 (Cb) outputs.
// Block = 128 threads = 4 waves; each wave computes a 32(M) x 64(N) slab
// (8 accumulators; B tiles reused across 2 M tiles -> 1.5 b128 loads per WMMA).
__global__ void gemm_bf16_nt(const uint16_t* __restrict__ A, const uint16_t* __restrict__ B,
                             const float* __restrict__ bias,
                             float* __restrict__ Cf, uint16_t* __restrict__ Cb,
                             int M, int N, int K) {
  int wave = threadIdx.x >> 5, lane = threadIdx.x & 31;
  int m0 = (blockIdx.y * 4 + wave) * 32;
  int n0 = blockIdx.x * 64;
  if (m0 >= M) return;
  v8f acc[2][4] = {};
  for (int k = 0; k < K; k += 32) {
    if (k + 64 < K) {  // prefetch 2 K-steps ahead (global_prefetch_b8)
      __builtin_prefetch(A + (size_t)m0 * K + k + 64, 0, 1);
      __builtin_prefetch(A + (size_t)(m0 + 16) * K + k + 64, 0, 1);
      __builtin_prefetch(B + (size_t)n0 * K + k + 64, 0, 1);
    }
    v16bf a0 = load_tile(A + (size_t)m0 * K + k, K);
    v16bf a1 = load_tile(A + (size_t)(m0 + 16) * K + k, K);
#pragma unroll
    for (int j = 0; j < 4; ++j) {
      v16bf b = load_tile(B + (size_t)(n0 + j * 16) * K + k, K);
      acc[0][j] = wmma_bf16(a0, b, acc[0][j]);
      acc[1][j] = wmma_bf16(a1, b, acc[1][j]);
    }
  }
  int n = lane & 15, mo = (lane >> 4) * 8;
#pragma unroll
  for (int i = 0; i < 2; ++i) {
#pragma unroll
    for (int j = 0; j < 4; ++j) {
      int col = n0 + j * 16 + n;
      float bv = bias ? bias[col] : 0.f;
#pragma unroll
      for (int r = 0; r < 8; ++r) {
        size_t idx = (size_t)(m0 + i * 16 + mo + r) * N + col;
        float v = acc[i][j][r] + bv;
        if (Cf) Cf[idx] = v;
        if (Cb) Cb[idx] = f2bf(v);
      }
    }
  }
}

// ---------------- Fused attention (one block per (b, head, 16-query tile)) ----------------
// 128 threads (4 waves). Dynamic LDS: S f32[16][T] | P bf16[16][T] | red f32[16][8]
// Emits ctx directly as bf16 (only consumed by the O-projection WMMA GEMM).
__global__ void k_attention(const uint16_t* __restrict__ Qb, const uint16_t* __restrict__ Kb,
                            const uint16_t* __restrict__ Vt, uint16_t* __restrict__ ctxb) {
  extern __shared__ char smem[];
  float*    S   = (float*)smem;                                   // 64 KB
  uint16_t* P   = (uint16_t*)(smem + 16 * T_SEQ * 4);             // 32 KB
  float*    red = (float*)(smem + 16 * T_SEQ * 4 + 16 * T_SEQ * 2);

  int blk = blockIdx.x;
  int qt = blk % (T_SEQ / 16); int rem = blk / (T_SEQ / 16);
  int hh = rem % NH_N; int b = rem / NH_N;
  int wave = threadIdx.x >> 5, lane = threadIdx.x & 31;

  const uint16_t* Qbase = Qb + ((size_t)(b * T_SEQ + qt * 16)) * H_DIM + hh * DK_N;

  // --- 1) S = Q K^T / sqrt(dk): 64 key tiles split across 4 waves ---
  for (int kt = wave; kt < T_SEQ / 16; kt += 4) {
    const uint16_t* Kbase = Kb + ((size_t)(b * T_SEQ + kt * 16)) * H_DIM + hh * DK_N;
    v8f acc = {};
#pragma unroll
    for (int kk = 0; kk < DK_N; kk += 32) {
      v16bf a  = load_tile(Qbase + kk, H_DIM);
      v16bf bb = load_tile(Kbase + kk, H_DIM);
      acc = wmma_bf16(a, bb, acc);
    }
    int n = lane & 15, mo = (lane >> 4) * 8;
#pragma unroll
    for (int r = 0; r < 8; ++r)
      S[(mo + r) * T_SEQ + kt * 16 + n] = acc[r] * 0.125f;        // 1/sqrt(64)
  }
  __syncthreads();

  // --- 2) softmax over 1024 keys: 16 rows x 8-thread groups ---
  {
    int row = threadIdx.x >> 3, sub = threadIdx.x & 7;
    float mx = -1e30f;
    for (int c = sub; c < T_SEQ; c += 8) mx = fmaxf(mx, S[row * T_SEQ + c]);
    red[row * 8 + sub] = mx;
    __syncthreads();
    float m2 = red[row * 8];
    for (int j = 1; j < 8; ++j) m2 = fmaxf(m2, red[row * 8 + j]);
    __syncthreads();
    float sm = 0.f;
    for (int c = sub; c < T_SEQ; c += 8) sm += __expf(S[row * T_SEQ + c] - m2);
    red[row * 8 + sub] = sm;
    __syncthreads();
    float tot = 0.f;
    for (int j = 0; j < 8; ++j) tot += red[row * 8 + j];
    float inv = 1.f / tot;
    for (int c = sub; c < T_SEQ; c += 8)
      P[row * T_SEQ + c] = f2bf(__expf(S[row * T_SEQ + c] - m2) * inv);
  }
  __syncthreads();

  // --- 3) ctx = P @ V^T: wave w handles dk tile w (64/16 = 4 tiles) ---
  {
    int nt = wave;
    const uint16_t* Vbase = Vt + ((size_t)((b * NH_N + hh) * DK_N + nt * 16)) * T_SEQ;
    v8f acc = {};
    for (int kc = 0; kc < T_SEQ; kc += 32) {
      v16bf a  = load_tile(P + kc, T_SEQ);        // from LDS
      v16bf bb = load_tile(Vbase + kc, T_SEQ);
      acc = wmma_bf16(a, bb, acc);
    }
    int n = lane & 15, mo = (lane >> 4) * 8;
#pragma unroll
    for (int r = 0; r < 8; ++r) {
      int q = qt * 16 + mo + r;
      ctxb[((size_t)(b * T_SEQ + q)) * H_DIM + hh * DK_N + nt * 16 + n] = f2bf(acc[r]);
    }
  }
}

// ---------------- LMU sequential recurrence (single persistent workgroup) ----------------
// 512 threads = 16 waves; h/m double-buffered in LDS as 16-row padded bf16 tiles.
// Per step: u[b] = ux + h.eh + m.em ; m' = m@A^T + u*Bd ; h' = tanh(Xp + h@Wh^T + m'@Wm^T)
__global__ void k_lmu_recurrence(const float* __restrict__ Xp,   // (B,T,H)
                                 const float* __restrict__ ux,   // (B,T)
                                 const float* __restrict__ eh,   // (H)
                                 const float* __restrict__ em,   // (MEM)
                                 const uint16_t* __restrict__ Whb, // (H,H)
                                 const uint16_t* __restrict__ Wmb, // (H,MEM)
                                 const uint16_t* __restrict__ Adb, // (MEM,MEM)
                                 const float* __restrict__ Bd,     // (MEM)
                                 float* __restrict__ ys,           // (B,T,H)
                                 float* __restrict__ hn,           // (B,H)
                                 float* __restrict__ mn) {         // (B,MEM)
  extern __shared__ char smem[];
  uint16_t* hbuf = (uint16_t*)smem;                                  // [2][16][H]
  uint16_t* mbuf = (uint16_t*)(smem + 2 * 16 * H_DIM * 2);           // [2][16][MEM]
  float*    u    = (float*)(smem + 2 * 16 * H_DIM * 2 + 2 * 16 * MEM_N * 2); // [16]

  int tid = threadIdx.x, lane = tid & 31, wave = tid >> 5;
  for (int i = tid; i < 2 * 16 * H_DIM; i += blockDim.x) hbuf[i] = 0;
  for (int i = tid; i < 2 * 16 * MEM_N; i += blockDim.x) mbuf[i] = 0;
  if (tid < 16) u[tid] = 0.f;
  __syncthreads();

  int cur = 0;
  for (int t = 0; t < T_SEQ; ++t) {
    uint16_t* hold = hbuf + cur * 16 * H_DIM;
    uint16_t* hnew = hbuf + (1 - cur) * 16 * H_DIM;
    uint16_t* mold = mbuf + cur * 16 * MEM_N;
    uint16_t* mnew = mbuf + (1 - cur) * 16 * MEM_N;

    // u[b]: waves 0..7, one batch row each, wave-reduced dot products
    if (wave < 8) {
      int b = wave;
      float s = 0.f;
      for (int c = lane; c < H_DIM; c += 32) s += bf2f(hold[b * H_DIM + c]) * eh[c];
      for (int c = lane; c < MEM_N; c += 32) s += bf2f(mold[b * MEM_N + c]) * em[c];
      for (int o = 16; o > 0; o >>= 1) s += __shfl_xor(s, o, 32);
      if (lane == 0) u[b] = s + ux[(size_t)b * T_SEQ + t];
    }
    __syncthreads();

    // m' = m @ A^T + u (x) Bd : 16 waves, one 16-col tile each
    {
      int nt = wave;
      v8f acc = {};
#pragma unroll
      for (int k = 0; k < MEM_N; k += 32) {
        v16bf a  = load_tile(mold + k, MEM_N);
        v16bf bb = load_tile(Adb + (size_t)nt * 16 * MEM_N + k, MEM_N);
        acc = wmma_bf16(a, bb, acc);
      }
      int n = lane & 15, mo = (lane >> 4) * 8;
      int col = nt * 16 + n;
#pragma unroll
      for (int r = 0; r < 8; ++r) {
        int row = mo + r;
        float v = acc[r] + ((row < B_SZ) ? u[row] * Bd[col] : 0.f);
        mnew[row * MEM_N + col] = f2bf(v);
        if (t == T_SEQ - 1 && row < B_SZ) mn[row * MEM_N + col] = v;
      }
    }
    __syncthreads();

    // h' = tanh(Xp[t] + h@Wh^T + m'@Wm^T): 16 waves x 2 tiles = 32 col tiles
#pragma unroll
    for (int j = 0; j < 2; ++j) {
      int nt = wave * 2 + j;
      v8f acc = {};
      for (int k = 0; k < H_DIM; k += 32) {
        if (k + 64 < H_DIM)
          __builtin_prefetch(Whb + (size_t)nt * 16 * H_DIM + k + 64, 0, 1);
        v16bf a  = load_tile(hold + k, H_DIM);
        v16bf bb = load_tile(Whb + (size_t)nt * 16 * H_DIM + k, H_DIM);
        acc = wmma_bf16(a, bb, acc);
      }
#pragma unroll
      for (int k = 0; k < MEM_N; k += 32) {
        v16bf a  = load_tile(mnew + k, MEM_N);
        v16bf bb = load_tile(Wmb + (size_t)nt * 16 * MEM_N + k, MEM_N);
        acc = wmma_bf16(a, bb, acc);
      }
      int n = lane & 15, mo = (lane >> 4) * 8;
      int col = nt * 16 + n;
#pragma unroll
      for (int r = 0; r < 8; ++r) {
        int row = mo + r;
        float v = 0.f;
        if (row < B_SZ) {
          v = tanhf(acc[r] + Xp[((size_t)row * T_SEQ + t) * H_DIM + col]);
          ys[((size_t)row * T_SEQ + t) * H_DIM + col] = v;
          if (t == T_SEQ - 1) hn[row * H_DIM + col] = v;
        }
        hnew[row * H_DIM + col] = f2bf(v);
      }
    }
    __syncthreads();
    cur ^= 1;
  }
}

// ---------------- Host orchestration ----------------
extern "C" void kernel_launch(void* const* d_in, const int* in_sizes, int n_in,
                              void* d_out, int out_size, void* d_ws, size_t ws_size,
                              hipStream_t stream) {
  (void)in_sizes; (void)n_in; (void)out_size; (void)ws_size;
  const float* x     = (const float*)d_in[0];
  const float* W_in  = (const float*)d_in[1];
  const float* b_in  = (const float*)d_in[2];
  const float* Wq    = (const float*)d_in[3];
  const float* bq    = (const float*)d_in[4];
  const float* Wk    = (const float*)d_in[5];
  const float* bk    = (const float*)d_in[6];
  const float* Wv    = (const float*)d_in[7];
  const float* bv    = (const float*)d_in[8];
  const float* Wo    = (const float*)d_in[9];
  const float* bo    = (const float*)d_in[10];
  const float* g_at  = (const float*)d_in[11];
  const float* b_at  = (const float*)d_in[12];
  const float* e_x   = (const float*)d_in[13];
  const float* e_h   = (const float*)d_in[14];
  const float* e_m   = (const float*)d_in[15];
  const float* Wx    = (const float*)d_in[16];
  const float* Wh    = (const float*)d_in[17];
  const float* Wm    = (const float*)d_in[18];
  const float* g_ln  = (const float*)d_in[19];
  const float* b_ln  = (const float*)d_in[20];
  const float* W_out = (const float*)d_in[21];
  const float* b_out = (const float*)d_in[22];
  float* out = (float*)d_out;

  // workspace carve-up (256B aligned)
  char* base = (char*)d_ws;
  size_t off = 0;
  auto carve = [&](size_t bytes) { char* p = base + off; off = (off + bytes + 255) & ~(size_t)255; return p; };
  uint16_t* xpad   = (uint16_t*)carve((size_t)BT * D_PAD * 2);
  uint16_t* winpad = (uint16_t*)carve((size_t)H_DIM * D_PAD * 2);
  uint16_t* wq_bf  = (uint16_t*)carve((size_t)H_DIM * H_DIM * 2);
  uint16_t* wk_bf  = (uint16_t*)carve((size_t)H_DIM * H_DIM * 2);
  uint16_t* wv_bf  = (uint16_t*)carve((size_t)H_DIM * H_DIM * 2);
  uint16_t* wo_bf  = (uint16_t*)carve((size_t)H_DIM * H_DIM * 2);
  uint16_t* wx_bf  = (uint16_t*)carve((size_t)H_DIM * H_DIM * 2);
  uint16_t* wh_bf  = (uint16_t*)carve((size_t)H_DIM * H_DIM * 2);
  uint16_t* wm_bf  = (uint16_t*)carve((size_t)H_DIM * MEM_N * 2);
  uint16_t* wout_b = (uint16_t*)carve((size_t)H_DIM * H_DIM * 2);
  uint16_t* ad_bf  = (uint16_t*)carve((size_t)MEM_N * MEM_N * 2);
  float*    bd_f   = (float*)carve(MEM_N * 4);
  const int EN = (MEM_N + 1) * (MEM_N + 1);
  float* eX = (float*)carve((size_t)EN * 4);
  float* eE = (float*)carve((size_t)EN * 4);
  float* eP = (float*)carve((size_t)EN * 4);
  float* eQ = (float*)carve((size_t)EN * 4);
  float*    h_f32 = (float*)carve((size_t)BT * H_DIM * 4);
  float*    bufA  = (float*)carve((size_t)BT * H_DIM * 4);
  float*    bufB  = (float*)carve((size_t)BT * H_DIM * 4);
  uint16_t* hx_bf = (uint16_t*)carve((size_t)BT * H_DIM * 2);
  uint16_t* q_bf  = (uint16_t*)carve((size_t)BT * H_DIM * 2);
  uint16_t* k_bf  = (uint16_t*)carve((size_t)BT * H_DIM * 2);
  uint16_t* vt_bf = (uint16_t*)carve((size_t)BT * H_DIM * 2);
  uint16_t* cx_bf = (uint16_t*)carve((size_t)BT * H_DIM * 2);
  float*    ux_f  = (float*)carve((size_t)BT * 4);

  auto g1 = [](int n) { return dim3((n + 255) / 256); };
  const dim3 blk256(256), blk128(128);
  const dim3 gemm_grid(H_DIM / 64, BT / 128);   // wave tile 32x64, 4 waves/block

  // ===== expm(M): scaling (2^-12) + 12-term Taylor + 12 squarings =====
  k_build_M<<<g1(EN), blk256, 0, stream>>>(eX, 1.f / 4096.f);
  k_eye_plus<<<g1(EN), blk256, 0, stream>>>(eE, eX);
  k_copy_f32<<<g1(EN), blk256, 0, stream>>>(eP, eX, EN);
  float* Pcur = eP; float* Pnxt = eQ;
  for (int k = 2; k <= 12; ++k) {
    k_gemm_naive<<<g1(EN), blk256, 0, stream>>>(Pcur, eX, Pnxt, 1.f / (float)k);
    k_add_f32<<<g1(EN), blk256, 0, stream>>>(eE, Pnxt, EN);
    float* t = Pcur; Pcur = Pnxt; Pnxt = t;
  }
  float* Ecur = eE; float* Enxt = Pnxt;
  for (int s = 0; s < 12; ++s) {
    k_gemm_naive<<<g1(EN), blk256, 0, stream>>>(Ecur, Ecur, Enxt, 1.f);
    float* t = Ecur; Ecur = Enxt; Enxt = t;
  }
  k_extract_AB<<<g1(MEM_N * MEM_N + MEM_N), blk256, 0, stream>>>(Ecur, ad_bf, bd_f);

  // ===== input projection: h = x @ W_in^T + b_in  (K padded 80 -> 96) =====
  k_pad_convert<<<g1(BT * D_PAD), blk256, 0, stream>>>(x, xpad, BT, D_IN, D_PAD);
  k_pad_convert<<<g1(H_DIM * D_PAD), blk256, 0, stream>>>(W_in, winpad, H_DIM, D_IN, D_PAD);
  gemm_bf16_nt<<<gemm_grid, blk128, 0, stream>>>(xpad, winpad, b_in, h_f32, nullptr, BT, H_DIM, D_PAD);

  const size_t attn_lds = 16 * T_SEQ * 4 + 16 * T_SEQ * 2 + 16 * 8 * 4; // ~96.5 KB
  const size_t lmu_lds  = 2 * 16 * H_DIM * 2 + 2 * 16 * MEM_N * 2 + 16 * 4;

  for (int l = 0; l < L_NUM; ++l) {
    const size_t wHH = (size_t)H_DIM * H_DIM, wHM = (size_t)H_DIM * MEM_N;
    k_f32_to_bf16<<<g1((int)wHH), blk256, 0, stream>>>(Wq + l * wHH, wq_bf, (int)wHH);
    k_f32_to_bf16<<<g1((int)wHH), blk256, 0, stream>>>(Wk + l * wHH, wk_bf, (int)wHH);
    k_f32_to_bf16<<<g1((int)wHH), blk256, 0, stream>>>(Wv + l * wHH, wv_bf, (int)wHH);
    k_f32_to_bf16<<<g1((int)wHH), blk256, 0, stream>>>(Wo + l * wHH, wo_bf, (int)wHH);
    k_f32_to_bf16<<<g1((int)wHH), blk256, 0, stream>>>(Wx + l * wHH, wx_bf, (int)wHH);
    k_f32_to_bf16<<<g1((int)wHH), blk256, 0, stream>>>(Wh + l * wHH, wh_bf, (int)wHH);
    k_f32_to_bf16<<<g1((int)wHM), blk256, 0, stream>>>(Wm + l * wHM, wm_bf, (int)wHM);

    // --- MHA ---
    k_f32_to_bf16<<<g1(BT * H_DIM), blk256, 0, stream>>>(h_f32, hx_bf, BT * H_DIM);
    // Q,K: bf16 straight out of the GEMM epilogue (no f32 round trip)
    gemm_bf16_nt<<<gemm_grid, blk128, 0, stream>>>(hx_bf, wq_bf, bq + l * H_DIM, nullptr, q_bf, BT, H_DIM, H_DIM);
    gemm_bf16_nt<<<gemm_grid, blk128, 0, stream>>>(hx_bf, wk_bf, bk + l * H_DIM, nullptr, k_bf, BT, H_DIM, H_DIM);
    // V: f32 out, then transpose-convert to (B,NH,DK,T) bf16
    gemm_bf16_nt<<<gemm_grid, blk128, 0, stream>>>(hx_bf, wv_bf, bv + l * H_DIM, bufA, nullptr, BT, H_DIM, H_DIM);
    k_transpose_v<<<g1(BT * H_DIM), blk256, 0, stream>>>(bufA, vt_bf);

    k_attention<<<dim3(B_SZ * NH_N * (T_SEQ / 16)), blk128, attn_lds, stream>>>(q_bf, k_bf, vt_bf, cx_bf);

    gemm_bf16_nt<<<gemm_grid, blk128, 0, stream>>>(cx_bf, wo_bf, bo + l * H_DIM, bufA, nullptr, BT, H_DIM, H_DIM);
    k_layernorm<<<dim3(BT / 8), blk256, 0, stream>>>(bufA, h_f32, g_at + l * H_DIM, b_at + l * H_DIM, h_f32, BT, H_DIM);

    // --- LMU: hoist Xp = xt@Wx^T and ux = xt@e_x, then sequential scan ---
    k_f32_to_bf16<<<g1(BT * H_DIM), blk256, 0, stream>>>(h_f32, hx_bf, BT * H_DIM);
    gemm_bf16_nt<<<gemm_grid, blk128, 0, stream>>>(hx_bf, wx_bf, nullptr, bufA, nullptr, BT, H_DIM, H_DIM);
    k_dot_rows<<<dim3(BT / 8), blk256, 0, stream>>>(h_f32, e_x + l * H_DIM, ux_f, BT, H_DIM);

    float* hn_out = out + (size_t)BT * H_DIM + (size_t)l * B_SZ * H_DIM;
    float* mn_out = out + (size_t)BT * H_DIM + (size_t)L_NUM * B_SZ * H_DIM + (size_t)l * B_SZ * MEM_N;
    k_lmu_recurrence<<<dim3(1), dim3(512), lmu_lds, stream>>>(
        bufA, ux_f, e_h + l * H_DIM, e_m + l * MEM_N,
        wh_bf, wm_bf, ad_bf, bd_f, bufB, hn_out, mn_out);

    k_layernorm<<<dim3(BT / 8), blk256, 0, stream>>>(bufB, nullptr, g_ln + l * H_DIM, b_ln + l * H_DIM, h_f32, BT, H_DIM);
  }

  // ===== final projection straight into d_out =====
  k_f32_to_bf16<<<g1(BT * H_DIM), blk256, 0, stream>>>(h_f32, hx_bf, BT * H_DIM);
  k_f32_to_bf16<<<g1((int)((size_t)H_DIM * H_DIM)), blk256, 0, stream>>>(W_out, wout_b, H_DIM * H_DIM);
  gemm_bf16_nt<<<gemm_grid, blk128, 0, stream>>>(hx_bf, wout_b, b_out, out, nullptr, BT, H_DIM, H_DIM);
}
